// SATCausalAttention_69827578299047
// MI455X (gfx1250) — matrix operand order repaired and checked by reading, not compiled, not run on hardware
//
#include <hip/hip_runtime.h>
#include <math.h>

// ---------------------------------------------------------------------------
// CDNA5 (gfx1250) causal attention forward, split-bf16 WMMA everywhere.
// Round 2: preload B-fragments per k-step and interleave WMMA chains across
// independent accumulators to fill WMMA->WMMA hazard slots (kill v_nops).
// ---------------------------------------------------------------------------

#define DEVFN static __device__ __forceinline__

typedef unsigned short u16;
typedef unsigned int   u32;
typedef __attribute__((ext_vector_type(8)))  u16    u16x8;
typedef __attribute__((ext_vector_type(16))) u16    u16x16;
typedef __attribute__((ext_vector_type(16))) __bf16 v16bf;
typedef __attribute__((ext_vector_type(8)))  float  v8f;

constexpr int BATCH  = 2;
constexpr int LSEQ   = 2048;
constexpr int DMODEL = 1024;
constexpr int NHEAD  = 16;
constexpr int DHEAD  = 64;
constexpr int MROWS  = BATCH * LSEQ;          // 4096
constexpr float NEGV = -1000000000.0f;

// ---- bf16 split helpers (bit-exact, RNE) ----------------------------------
DEVFN u16 f2bf(float f) {
  u32 u = __builtin_bit_cast(u32, f);
  u32 r = u + 0x7fffu + ((u >> 16) & 1u);
  return (u16)(r >> 16);
}
DEVFN float bf2f(u16 h) {
  u32 u = ((u32)h) << 16;
  return __builtin_bit_cast(float, u);
}

DEVFN v8f vzero8() { v8f z = {0.f,0.f,0.f,0.f,0.f,0.f,0.f,0.f}; return z; }

// ---- WMMA wrapper: D = A(bf16) x B(bf16) + C(f32) -------------------------
DEVFN v8f wmma_bf16(u16x16 a, u16x16 b, v8f c) {
  return __builtin_amdgcn_wmma_f32_16x16x32_bf16(
      false, __builtin_bit_cast(v16bf, a),
      false, __builtin_bit_cast(v16bf, b),
      (short)0, c, false, false);
}

// A-fragment 16x32: lane holds row m; element e -> k = g*16 + e
// -> two contiguous 8-element (16B) loads at offsets g*16 and g*16+8.
DEVFN u16x16 load_afrag(const u16* row, int g) {
  u16x8 a0 = *(const u16x8*)(row + g * 16);
  u16x8 a1 = *(const u16x8*)(row + g * 16 + 8);
  u16x16 r;
#pragma unroll
  for (int i = 0; i < 8; i++) { r[i] = a0[i]; r[8 + i] = a1[i]; }
  return r;
}
// B-fragment 32x16: lane holds col n; element e -> k = g*16 + e
// -> one contiguous 16-element (32B) load; caller passes base + g*16.
DEVFN u16x16 load_bfrag(const u16* p) {
  u16x8 a0 = *(const u16x8*)(p);
  u16x8 a1 = *(const u16x8*)(p + 8);
  u16x16 r;
#pragma unroll
  for (int i = 0; i < 8; i++) { r[i] = a0[i]; r[8 + i] = a1[i]; }
  return r;
}

// ---------------------------------------------------------------------------
// Kernel 1: elementwise fp32 -> (bf16 hi, bf16 lo)
// ---------------------------------------------------------------------------
__global__ void split_kernel(const float* __restrict__ src,
                             u16* __restrict__ hi, u16* __restrict__ lo, int n) {
  int i = blockIdx.x * blockDim.x + threadIdx.x;
  if (i < n) {
    float f = src[i];
    u16 h = f2bf(f);
    hi[i] = h;
    lo[i] = f2bf(f - bf2f(h));
  }
}

// ---------------------------------------------------------------------------
// Kernel 2: W[KxN] fp32 -> WT[NxK] bf16 hi/lo  (so B-frags are contiguous)
// ---------------------------------------------------------------------------
__global__ void split_transpose_kernel(const float* __restrict__ W,
                                       u16* __restrict__ hi, u16* __restrict__ lo,
                                       int K, int N) {
  int i = blockIdx.x * blockDim.x + threadIdx.x;
  if (i < K * N) {
    int k = i / N, n = i - k * N;
    float f = W[i];
    u16 h = f2bf(f);
    size_t o = (size_t)n * K + k;
    hi[o] = h;
    lo[o] = f2bf(f - bf2f(h));
  }
}

// ---------------------------------------------------------------------------
// Kernel 3: generic split-bf16 GEMM, C = A[MxK] * B[KxN] (+bias)(*scale)
//   B supplied transposed: BT[NxK].  Per wave: 16x64 output tile.
//   mode 0: fp32 row-major to outF
//   mode 1: bf16 split to [b][h][l][d]   (Q/K layout)
//   mode 2: bf16 split to [b][h][d][l]   (V transposed layout)
// ---------------------------------------------------------------------------
__global__ __launch_bounds__(256) void gemm_split_bf16(
    const u16* __restrict__ Ahi, const u16* __restrict__ Alo,
    const u16* __restrict__ BThi, const u16* __restrict__ BTlo,
    const float* __restrict__ bias, const float* __restrict__ scale_ptr,
    float* __restrict__ outF, u16* __restrict__ outHi, u16* __restrict__ outLo,
    int M, int K, int N, int mode) {
  const int wave = threadIdx.x >> 5;
  const int lane = threadIdx.x & 31;
  const int ln = lane & 15, g = lane >> 4;
  const int row0 = blockIdx.y * 128 + wave * 16;
  const int col0 = blockIdx.x * 64;

  v8f acc[4];
#pragma unroll
  for (int t = 0; t < 4; t++) acc[t] = vzero8();

  const u16* arh = Ahi + (size_t)(row0 + ln) * K;
  const u16* arl = Alo + (size_t)(row0 + ln) * K;

  for (int kk = 0; kk < K; kk += 32) {
    u16x16 ah = load_afrag(arh + kk, g);
    u16x16 al = load_afrag(arl + kk, g);
    u16x16 Bh[4], Bl[4];
#pragma unroll
    for (int nt = 0; nt < 4; nt++) {
      const int n = col0 + nt * 16 + ln;
      Bh[nt] = load_bfrag(BThi + (size_t)n * K + kk + g * 16);
      Bl[nt] = load_bfrag(BTlo + (size_t)n * K + kk + g * 16);
    }
    if (kk + 32 < K) {  // hint the next A tile into cache (global_prefetch_b8)
      __builtin_prefetch(arh + kk + 32, 0, 1);
      __builtin_prefetch(arl + kk + 32, 0, 1);
    }
    // Three interleaved passes over 4 independent accumulators (4-way ILP)
#pragma unroll
    for (int nt = 0; nt < 4; nt++) acc[nt] = wmma_bf16(al, Bh[nt], acc[nt]);
#pragma unroll
    for (int nt = 0; nt < 4; nt++) acc[nt] = wmma_bf16(ah, Bl[nt], acc[nt]);
#pragma unroll
    for (int nt = 0; nt < 4; nt++) acc[nt] = wmma_bf16(ah, Bh[nt], acc[nt]);
  }

  const float scale = scale_ptr ? *scale_ptr : 1.0f;
#pragma unroll
  for (int nt = 0; nt < 4; nt++) {
    const int c = col0 + nt * 16 + ln;
    const float bs = bias ? bias[c] : 0.0f;
#pragma unroll
    for (int v = 0; v < 8; v++) {
      const int r = row0 + v + 8 * g;       // C/D layout: row = v + 8*g
      float x = (acc[nt][v] + bs) * scale;
      if (mode == 0) {
        outF[(size_t)r * N + c] = x;
      } else {
        int b = r >> 11, l = r & (LSEQ - 1);
        int h = c >> 6, d = c & (DHEAD - 1);
        size_t idx = (mode == 1)
            ? ((size_t)((b * NHEAD + h) * LSEQ + l) * DHEAD + d)
            : ((size_t)((b * NHEAD + h) * DHEAD + d) * LSEQ + l);
        u16 hb = f2bf(x);
        outHi[idx] = hb;
        outLo[idx] = f2bf(x - bf2f(hb));
      }
    }
  }
}

// ---------------------------------------------------------------------------
// Kernel 4: fused causal attention. One wave = one (b,h) x 16 query rows.
// Pass A: streaming online softmax stats (WMMA scores, two dt-chains).
// Pass B: recompute scores, write exact probs to d_out, P*V via WMMA
//         (prob tile transposed D-layout -> A-layout through LDS).
// ---------------------------------------------------------------------------
__global__ __launch_bounds__(128) void attn_fused(
    const u16* __restrict__ Qhi, const u16* __restrict__ Qlo,
    const u16* __restrict__ Khi, const u16* __restrict__ Klo,
    const u16* __restrict__ Vthi, const u16* __restrict__ Vtlo,
    float* __restrict__ attn_out, u16* __restrict__ ctxHi, u16* __restrict__ ctxLo) {
  __shared__ float ldsP[4][16][33];  // per-wave 16(q) x 32(k) staging (+pad)

  const int wid = threadIdx.x >> 5;
  const int lane = threadIdx.x & 31;
  const int ln = lane & 15, g = lane >> 4;
  const int w = blockIdx.x * 4 + wid;   // global wave id
  const int bh = w >> 7;                // 128 q-tiles per (b,h)
  const int qt = w & 127;
  const int q0 = qt * 16;

  const size_t headQK = (size_t)bh * LSEQ * DHEAD;
  const u16* kh_ = Khi + headQK;
  const u16* kl_ = Klo + headQK;
  const u16* vth = Vthi + headQK;       // [DHEAD][LSEQ]
  const u16* vtl = Vtlo + headQK;

  // Q fragments for dt = 0,1 halves of DHEAD (pre-scaled by 1/sqrt(dh))
  u16x16 qfh[2], qfl[2];
  {
    const u16* qr_h = Qhi + headQK + (size_t)(q0 + ln) * DHEAD;
    const u16* qr_l = Qlo + headQK + (size_t)(q0 + ln) * DHEAD;
#pragma unroll
    for (int dt = 0; dt < 2; dt++) {
      qfh[dt] = load_afrag(qr_h + dt * 32, g);
      qfl[dt] = load_afrag(qr_l + dt * 32, g);
    }
  }

  // ---- pass A: online max / sum over unmasked key tiles 0..qt ----
  float mrow[8], lrow[8];
#pragma unroll
  for (int v = 0; v < 8; v++) { mrow[v] = -1e30f; lrow[v] = 0.0f; }

  for (int kt = 0; kt <= qt; kt++) {
    const u16* kr_h = kh_ + (size_t)(kt * 16 + ln) * DHEAD;
    const u16* kr_l = kl_ + (size_t)(kt * 16 + ln) * DHEAD;
    u16x16 Bh[2], Bl[2];
#pragma unroll
    for (int dt = 0; dt < 2; dt++) {
      Bh[dt] = load_bfrag(kr_h + dt * 32 + g * 16);
      Bl[dt] = load_bfrag(kr_l + dt * 32 + g * 16);
    }
    // two independent 3-chains (one per dt), summed afterwards
    v8f sa = vzero8(), sb = vzero8();
    sa = wmma_bf16(qfl[0], Bh[0], sa);  sb = wmma_bf16(qfl[1], Bh[1], sb);
    sa = wmma_bf16(qfh[0], Bl[0], sa);  sb = wmma_bf16(qfh[1], Bl[1], sb);
    sa = wmma_bf16(qfh[0], Bh[0], sa);  sb = wmma_bf16(qfh[1], Bh[1], sb);
    v8f s = sa + sb;
#pragma unroll
    for (int v = 0; v < 8; v++) {
      float t = s[v];
      if (kt == qt) {  // diagonal tile: causal mask
        int q = q0 + v + 8 * g;
        int k = kt * 16 + ln;
        if (k > q) t = NEGV;
      }
      float tm = t;
      tm = fmaxf(tm, __shfl_xor(tm, 1, 16));
      tm = fmaxf(tm, __shfl_xor(tm, 2, 16));
      tm = fmaxf(tm, __shfl_xor(tm, 4, 16));
      tm = fmaxf(tm, __shfl_xor(tm, 8, 16));
      float mnew = fmaxf(mrow[v], tm);
      float p = __expf(t - mnew);
      float ps = p;
      ps += __shfl_xor(ps, 1, 16);
      ps += __shfl_xor(ps, 2, 16);
      ps += __shfl_xor(ps, 4, 16);
      ps += __shfl_xor(ps, 8, 16);
      lrow[v] = lrow[v] * __expf(mrow[v] - mnew) + ps;
      mrow[v] = mnew;
    }
  }
  float invl[8];
#pragma unroll
  for (int v = 0; v < 8; v++) invl[v] = 1.0f / lrow[v];

  // ---- pass B: probs -> d_out, P*V -> ctx ----
  float* arow0 = attn_out + ((size_t)bh * LSEQ + q0) * LSEQ;
  v8f ctx[4];
#pragma unroll
  for (int t = 0; t < 4; t++) ctx[t] = vzero8();

  for (int ks = 0; ks < LSEQ / 32; ks++) {
    const int kk = ks * 32;
    if (kk > q0 + 15) {  // fully masked 32-key step: exact zeros
#pragma unroll
      for (int v = 0; v < 8; v++) {
        float* pr = arow0 + (size_t)(v + 8 * g) * LSEQ + kk;
        pr[ln] = 0.0f;
        pr[16 + ln] = 0.0f;
      }
      continue;
    }
    v8f s0 = vzero8(), s1 = vzero8();
    {
      const u16* k0h = kh_ + (size_t)(kk + ln) * DHEAD;
      const u16* k0l = kl_ + (size_t)(kk + ln) * DHEAD;
      const u16* k1h = kh_ + (size_t)(kk + 16 + ln) * DHEAD;
      const u16* k1l = kl_ + (size_t)(kk + 16 + ln) * DHEAD;
      u16x16 B0h[2], B0l[2], B1h[2], B1l[2];
#pragma unroll
      for (int dt = 0; dt < 2; dt++) {
        B0h[dt] = load_bfrag(k0h + dt * 32 + g * 16);
        B0l[dt] = load_bfrag(k0l + dt * 32 + g * 16);
        B1h[dt] = load_bfrag(k1h + dt * 32 + g * 16);
        B1l[dt] = load_bfrag(k1l + dt * 32 + g * 16);
      }
      // interleave the two independent score chains
      s0 = wmma_bf16(qfl[0], B0h[0], s0);  s1 = wmma_bf16(qfl[0], B1h[0], s1);
      s0 = wmma_bf16(qfl[1], B0h[1], s0);  s1 = wmma_bf16(qfl[1], B1h[1], s1);
      s0 = wmma_bf16(qfh[0], B0l[0], s0);  s1 = wmma_bf16(qfh[0], B1l[0], s1);
      s0 = wmma_bf16(qfh[1], B0l[1], s0);  s1 = wmma_bf16(qfh[1], B1l[1], s1);
      s0 = wmma_bf16(qfh[0], B0h[0], s0);  s1 = wmma_bf16(qfh[0], B1h[0], s1);
      s0 = wmma_bf16(qfh[1], B0h[1], s0);  s1 = wmma_bf16(qfh[1], B1h[1], s1);
    }
    // probs, attn store, LDS staging (D-layout)
#pragma unroll
    for (int v = 0; v < 8; v++) {
      const int m = v + 8 * g;
      const int q = q0 + m;
      const int k0i = kk + ln, k1i = kk + 16 + ln;
      float p0 = (k0i > q) ? 0.0f : __expf(s0[v] - mrow[v]) * invl[v];
      float p1 = (k1i > q) ? 0.0f : __expf(s1[v] - mrow[v]) * invl[v];
      float* pr = arow0 + (size_t)m * LSEQ + kk;
      pr[ln] = p0;
      pr[16 + ln] = p1;
      ldsP[wid][m][ln] = p0;
      ldsP[wid][m][16 + ln] = p1;
    }
    asm volatile("s_wait_dscnt 0" ::: "memory");  // LDS write -> read (same wave)
    // rebuild probs as split-bf16 A-fragment (lane = q row, k along elements)
    u16x16 pah, pal;
#pragma unroll
    for (int e = 0; e < 16; e++) {
      int kloc = g * 16 + e;
      float f = ldsP[wid][ln][kloc];
      u16 hb = f2bf(f);
      pah[e] = hb;
      pal[e] = f2bf(f - bf2f(hb));
    }
    u16x16 Vh[4], Vl[4];
#pragma unroll
    for (int dsub = 0; dsub < 4; dsub++) {
      Vh[dsub] = load_bfrag(vth + (size_t)(dsub * 16 + ln) * LSEQ + kk + g * 16);
      Vl[dsub] = load_bfrag(vtl + (size_t)(dsub * 16 + ln) * LSEQ + kk + g * 16);
    }
    // three interleaved passes over 4 independent ctx accumulators
#pragma unroll
    for (int dsub = 0; dsub < 4; dsub++) ctx[dsub] = wmma_bf16(pal, Vh[dsub], ctx[dsub]);
#pragma unroll
    for (int dsub = 0; dsub < 4; dsub++) ctx[dsub] = wmma_bf16(pah, Vl[dsub], ctx[dsub]);
#pragma unroll
    for (int dsub = 0; dsub < 4; dsub++) ctx[dsub] = wmma_bf16(pah, Vh[dsub], ctx[dsub]);
  }

  // epilogue: ctx -> bf16 split, row-major [b*L + q][h*64 + d] for final GEMM
  const int b = bh >> 4, h = bh & 15;
#pragma unroll
  for (int dsub = 0; dsub < 4; dsub++) {
#pragma unroll
    for (int v = 0; v < 8; v++) {
      const int q = q0 + v + 8 * g;
      const int d = dsub * 16 + ln;
      size_t idx = ((size_t)(b * LSEQ + q)) * DMODEL + h * DHEAD + d;
      float x = ctx[dsub][v];
      u16 hb = f2bf(x);
      ctxHi[idx] = hb;
      ctxLo[idx] = f2bf(x - bf2f(hb));
    }
  }
}

// ---------------------------------------------------------------------------
// Host orchestration
// ---------------------------------------------------------------------------
extern "C" void kernel_launch(void* const* d_in, const int* in_sizes, int n_in,
                              void* d_out, int out_size, void* d_ws, size_t ws_size,
                              hipStream_t stream) {
  (void)in_sizes; (void)n_in; (void)out_size; (void)ws_size;
  const float* x     = (const float*)d_in[0];
  // d_in[1] = attn_mask (triu k=1) -> applied analytically
  const float* Wq    = (const float*)d_in[2];
  const float* bq    = (const float*)d_in[3];
  const float* Wk    = (const float*)d_in[4];
  const float* bk    = (const float*)d_in[5];
  const float* Wv    = (const float*)d_in[6];
  const float* bv    = (const float*)d_in[7];
  const float* Wo    = (const float*)d_in[8];
  const float* bo    = (const float*)d_in[9];
  const float* scale = (const float*)d_in[10];

  float* outp = (float*)d_out;                                  // [B,L,DMODEL]
  float* attn = outp + (size_t)BATCH * LSEQ * DMODEL;           // [B,H,L,L]

  const size_t XE = (size_t)MROWS * DMODEL;    // 4,194,304 elems
  const size_t WE = (size_t)DMODEL * DMODEL;   // 1,048,576 elems

  u16* p = (u16*)d_ws;
  u16 *xhi = p; p += XE; u16 *xlo = p; p += XE;   // reused as ctx hi/lo later
  u16 *wqh = p; p += WE; u16 *wql = p; p += WE;
  u16 *wkh = p; p += WE; u16 *wkl = p; p += WE;
  u16 *wvh = p; p += WE; u16 *wvl = p; p += WE;
  u16 *woh = p; p += WE; u16 *wol = p; p += WE;
  u16 *q_h = p; p += XE; u16 *q_l = p; p += XE;
  u16 *k_h = p; p += XE; u16 *k_l = p; p += XE;
  u16 *vth = p; p += XE; u16 *vtl = p; p += XE;   // ~80 MB total

  dim3 blk(256);
  split_kernel<<<dim3((unsigned)((XE + 255) / 256)), blk, 0, stream>>>(x, xhi, xlo, (int)XE);
  split_transpose_kernel<<<dim3((unsigned)((WE + 255) / 256)), blk, 0, stream>>>(Wq, wqh, wql, DMODEL, DMODEL);
  split_transpose_kernel<<<dim3((unsigned)((WE + 255) / 256)), blk, 0, stream>>>(Wk, wkh, wkl, DMODEL, DMODEL);
  split_transpose_kernel<<<dim3((unsigned)((WE + 255) / 256)), blk, 0, stream>>>(Wv, wvh, wvl, DMODEL, DMODEL);
  split_transpose_kernel<<<dim3((unsigned)((WE + 255) / 256)), blk, 0, stream>>>(Wo, woh, wol, DMODEL, DMODEL);

  dim3 ggrid(DMODEL / 64, MROWS / 128);  // (16, 32), 8 waves/block, 16x64/wave
  gemm_split_bf16<<<ggrid, blk, 0, stream>>>(xhi, xlo, wqh, wql, bq, scale,
                                             nullptr, q_h, q_l, MROWS, DMODEL, DMODEL, 1);
  gemm_split_bf16<<<ggrid, blk, 0, stream>>>(xhi, xlo, wkh, wkl, bk, nullptr,
                                             nullptr, k_h, k_l, MROWS, DMODEL, DMODEL, 1);
  gemm_split_bf16<<<ggrid, blk, 0, stream>>>(xhi, xlo, wvh, wvl, bv, nullptr,
                                             nullptr, vth, vtl, MROWS, DMODEL, DMODEL, 2);

  // 32 (b,h) * 128 q-tiles = 4096 waves = 1024 blocks of 4 waves
  attn_fused<<<dim3(1024), dim3(128), 0, stream>>>(q_h, q_l, k_h, k_l, vth, vtl,
                                                   attn, xhi, xlo);

  gemm_split_bf16<<<ggrid, blk, 0, stream>>>(xhi, xlo, woh, wol, bo, nullptr,
                                             outp, nullptr, nullptr, MROWS, DMODEL, DMODEL, 0);
}
